// MultiHeadAttention_15977278341984
// MI455X (gfx1250) — matrix operand order repaired and checked by reading, compiled, not run on hardware
//
#include <hip/hip_runtime.h>
#include <hip/hip_bf16.h>
#include <cstdint>
#include <cstddef>

// ---------------------------------------------------------------------------
// MI455X (gfx1250) fused multi-head attention, compile-only target.
// GEMMs: v_wmma_f32_16x16x32_f16 (wave32 WMMA, f32 accumulate).
// f16 tile staging: global_load_async_to_lds_b128 (ASYNCcnt path).
// ---------------------------------------------------------------------------

typedef __attribute__((ext_vector_type(16))) _Float16 v16h;
typedef __attribute__((ext_vector_type(8)))  _Float16 v8h;
typedef __attribute__((ext_vector_type(8)))  float    v8f;

#define B_   8
#define S_   1024
#define DM_  512
#define H_   8
#define DK_  64
#define M_   8192      // B_*S_

union V16U { v16h v; v8h h[2]; };

// A-matrix fragment (16x32 f16): lane holds row m=lane&15; K chunks at
// kb=(lane>>4)*8 and kb+16 (ISA 7.12.2 16-bit A layout). Two 16B LDS reads.
__device__ __forceinline__ v16h ldsfragA(const _Float16* p) {
  V16U u;
  u.h[0] = *(const v8h*)(p);
  u.h[1] = *(const v8h*)(p + 16);
  return u.v;
}
// B-matrix fragment (32x16 f16): lane holds col n=lane&15 (row of B^T);
// K chunks at kb=(lane>>4)*8 and kb+16 (same order as A). Two 16B LDS reads.
__device__ __forceinline__ v16h ldsfragB(const _Float16* p) {
  V16U u;
  u.h[0] = *(const v8h*)(p);
  u.h[1] = *(const v8h*)(p + 16);
  return u.v;
}

__device__ __forceinline__ v8f wmma_f16(v16h a, v16h b, v8f c) {
  return __builtin_amdgcn_wmma_f32_16x16x32_f16(
      /*neg_a=*/false, a, /*neg_b=*/false, b,
      /*c_mod=*/(short)0, c, /*reuse_a=*/false, /*reuse_b=*/false);
}

// CDNA5 async memory->LDS copy (16B per active lane), tracked by ASYNCcnt.
// LDS byte offset = low 32 bits of the generic shared-space address
// (flat aperture rule).
__device__ __forceinline__ void async_load_b128(void* ldsDst, const void* gsrc) {
  asm volatile("global_load_async_to_lds_b128 %0, %1, off"
               :: "v"((unsigned)(uintptr_t)ldsDst), "v"(gsrc)
               : "memory");
}
__device__ __forceinline__ void wait_async0() {
  asm volatile("s_wait_asynccnt 0x0" ::: "memory");
}

// ---------------------------------------------------------------------------
// Kernel 1: convert 4 weight matrices f32 [512][512] -> f16 transposed
// Wt[mat][n*512 + k] = (f16) W[mat][k*512 + n]
// ---------------------------------------------------------------------------
__global__ __launch_bounds__(256)
void cvt_weights(const float* __restrict__ Wq, const float* __restrict__ Wk,
                 const float* __restrict__ Wv, const float* __restrict__ Wfc,
                 _Float16* __restrict__ WtBase) {
  const int idx = blockIdx.x * 256 + threadIdx.x;   // 0 .. 4*262144-1
  const int mat = idx >> 18;
  const int e   = idx & 262143;
  const int nn  = e >> 9;
  const int kk  = e & 511;
  const float* srcs[4] = {Wq, Wk, Wv, Wfc};
  WtBase[idx] = (_Float16)srcs[mat][kk * DM_ + nn];
}

// ---------------------------------------------------------------------------
// Kernel 2: C[8192,512] = A(f32)[8192,512] @ W  where Bt = W^T (f16, [512][512]).
// 64x64 block tile, 4 waves, each wave a 32x32 quadrant = 4 WMMA / k-step.
// B^T tile staged via async-to-LDS; A tile batched f32 loads -> f16 LDS.
// mode 0: Cout f16 row-major [8192][512]   (Q, K)
// mode 1: Cout = Vt f16 [(b*8+h)*64+d][1024]  (transposed V store)
// ---------------------------------------------------------------------------
__global__ __launch_bounds__(128)
void gemm_xw(const float* __restrict__ A, const _Float16* __restrict__ Bt,
             _Float16* __restrict__ Cout, int mode) {
  __shared__ _Float16 As[64 * 40];   // 64 rows x 32 k-halves, stride 40
  __shared__ _Float16 Bs[64 * 40];
  const int tid  = threadIdx.x;
  const int lane = tid & 31, w = tid >> 5;
  const int bm = blockIdx.y * 64, bn = blockIdx.x * 64;
  const int wr = (w >> 1) * 32, wc = (w & 1) * 32;
  const int m = lane & 15, hl = lane >> 4;
  const int kbA = hl * 8, kbB = hl * 8;
  v8f acc[2][2] = {};

  for (int kt = 0; kt < DM_; kt += 32) {
    __syncthreads();
    // B^T tile: 64x32 f16, async memory->LDS (128 thr x 2 x 16B)
#pragma unroll
    for (int j = 0; j < 2; ++j) {
      const int idx = tid + j * 128;
      const int r = idx >> 2, c8 = (idx & 3) * 8;
      async_load_b128(Bs + r * 40 + c8,
                      Bt + (size_t)(bn + r) * DM_ + kt + c8);
    }
    // A tile: 64x32 f32 -> f16; batch all loads first for MLP
    float4 fa[4];
#pragma unroll
    for (int j = 0; j < 4; ++j) {
      const int idx = tid + j * 128;
      const int r = idx >> 3, c4 = (idx & 7) * 4;
      fa[j] = *(const float4*)(A + (size_t)(bm + r) * DM_ + kt + c4);
    }
#pragma unroll
    for (int j = 0; j < 4; ++j) {
      const int idx = tid + j * 128;
      const int r = idx >> 3, c4 = (idx & 7) * 4;
      _Float16* d = As + r * 40 + c4;
      d[0] = (_Float16)fa[j].x; d[1] = (_Float16)fa[j].y;
      d[2] = (_Float16)fa[j].z; d[3] = (_Float16)fa[j].w;
    }
    if (kt + 32 < DM_) {  // gfx1250 global_prefetch_b8 of next A tile
      __builtin_prefetch(A + (size_t)(bm + (tid >> 3)) * DM_ + kt + 32, 0, 1);
    }
    wait_async0();
    __syncthreads();
    const v16h a0 = ldsfragA(As + (wr +  0 + m) * 40 + kbA);
    const v16h a1 = ldsfragA(As + (wr + 16 + m) * 40 + kbA);
    const v16h b0 = ldsfragB(Bs + (wc +  0 + m) * 40 + kbB);
    const v16h b1 = ldsfragB(Bs + (wc + 16 + m) * 40 + kbB);
    acc[0][0] = wmma_f16(a0, b0, acc[0][0]);
    acc[0][1] = wmma_f16(a0, b1, acc[0][1]);
    acc[1][0] = wmma_f16(a1, b0, acc[1][0]);
    acc[1][1] = wmma_f16(a1, b1, acc[1][1]);
  }

  const int n = m;
#pragma unroll
  for (int ti = 0; ti < 2; ++ti)
#pragma unroll
    for (int tj = 0; tj < 2; ++tj) {
      const int r0 = bm + wr + ti * 16;
      const int c0 = bn + wc + tj * 16;
#pragma unroll
      for (int r = 0; r < 8; ++r) {
        const int gm = r0 + r + 8 * hl;     // C layout: row = vgpr + 8*(lane>>4)
        const int gn = c0 + n;              //           col = lane&15
        const _Float16 hv = (_Float16)acc[ti][tj][r];
        if (mode == 0) {
          Cout[(size_t)gm * DM_ + gn] = hv;
        } else {
          const int b = gm >> 10, s = gm & 1023;
          const int h = gn >> 6,  d = gn & 63;
          Cout[(((size_t)((b * H_ + h) * DK_ + d)) << 10) | (size_t)s] = hv;
        }
      }
    }
}

// ---------------------------------------------------------------------------
// Kernel 3: fused scores + mask + softmax.
// Grid (64 q-tiles, 64 bh). Block 128 (4 waves). Each wave owns 16 key-tiles.
// scores[16][1024] f32 kept in dynamic LDS; normalized attn -> d_out.
// Q/K tiles staged via async-to-LDS.
// ---------------------------------------------------------------------------
#define ATTN_SMEM (65536 + 2304 + 9216 + 512 + 64)

__global__ __launch_bounds__(128)
void attn_softmax(const _Float16* __restrict__ Qf, const _Float16* __restrict__ Kf,
                  const unsigned char* __restrict__ mask, float* __restrict__ attn) {
  extern __shared__ char smem[];
  float*    sc   = (float*)smem;                               // [16][1024]
  _Float16* Qt   = (_Float16*)(smem + 65536);                  // [16][72]
  _Float16* Kt   = Qt + 16 * 72;                               // [4][16][72]
  float*    red  = (float*)(smem + 65536 + 2304 + 9216);       // [16][8]
  float*    rowv = red + 128;                                  // [16]

  const int tid = threadIdx.x;
  const int lane = tid & 31, w = tid >> 5;
  const int qbase = blockIdx.x * 16;
  const int bh = blockIdx.y;
  const int b = bh >> 3, h = bh & 7;

  // Load 16x64 Q tile (f16) into LDS via async copy
  {
    const int r = tid >> 3, c8 = (tid & 7) * 8;
    async_load_b128(Qt + r * 72 + c8,
                    Qf + (size_t)(b * S_ + qbase + r) * DM_ + h * DK_ + c8);
  }
  wait_async0();
  __syncthreads();

  const int m = lane & 15, hl = lane >> 4;
  const int kbA = hl * 8, kbB = hl * 8;
  const v16h aq0 = ldsfragA(Qt + m * 72 +  0 + kbA);
  const v16h aq1 = ldsfragA(Qt + m * 72 + 32 + kbA);
  _Float16* Ktw = Kt + w * 16 * 72;
  const int n = m;

  for (int i = 0; i < 16; ++i) {
    const int kt = w * 16 + i;
    // This wave stages its 16x64 K tile into its private LDS region (async)
#pragma unroll
    for (int j = 0; j < 4; ++j) {
      const int idx = lane + 32 * j;
      const int r = idx >> 3, c8 = (idx & 7) * 8;
      async_load_b128(Ktw + r * 72 + c8,
                      Kf + (size_t)(b * S_ + kt * 16 + r) * DM_ + h * DK_ + c8);
    }
    wait_async0();  // per-wave: our async writes are now visible in LDS
    const v16h bk0 = ldsfragB(Ktw + m * 72 +  0 + kbB);
    const v16h bk1 = ldsfragB(Ktw + m * 72 + 32 + kbB);
    v8f c = {};
    c = wmma_f16(aq0, bk0, c);
    c = wmma_f16(aq1, bk1, c);
#pragma unroll
    for (int r = 0; r < 8; ++r) {
      const int q = qbase + r + 8 * hl;
      const int k = kt * 16 + n;
      float v = c[r] * 0.125f;  // 1/sqrt(64)
      if (mask[(size_t)b * (S_ * (size_t)S_) + (size_t)q * S_ + k]) v = -1e9f;
      sc[(r + 8 * hl) * S_ + k] = v;
    }
  }
  __syncthreads();

  // Softmax: 8 threads per row
  const int row = tid >> 3, sub = tid & 7;
  float* srow = sc + row * S_;
  float lmax = -3.0e38f;
  for (int c = sub; c < S_; c += 8) lmax = fmaxf(lmax, srow[c]);
  red[row * 8 + sub] = lmax;
  __syncthreads();
  if (sub == 0) {
    float mx = red[row * 8];
#pragma unroll
    for (int j = 1; j < 8; ++j) mx = fmaxf(mx, red[row * 8 + j]);
    rowv[row] = mx;
  }
  __syncthreads();
  const float mx = rowv[row];
  float lsum = 0.f;
  for (int c = sub; c < S_; c += 8) {
    const float e = __expf(srow[c] - mx);
    srow[c] = e;
    lsum += e;
  }
  __syncthreads();
  red[row * 8 + sub] = lsum;
  __syncthreads();
  if (sub == 0) {
    float s = 0.f;
#pragma unroll
    for (int j = 0; j < 8; ++j) s += red[row * 8 + j];
    rowv[row] = s;
  }
  __syncthreads();
  const float inv = 1.0f / rowv[row];
  float* dst = attn + ((size_t)bh * S_ + qbase + row) * S_;
  for (int c = sub; c < S_; c += 8) dst[c] = srow[c] * inv;
}

// ---------------------------------------------------------------------------
// Kernel 4: context = attn(f32, from d_out) @ V, via Vt f16 [(bh)*64+d][1024].
// Grid (16 q-tiles of 64, 64 bh). Each wave: 16 q-rows x 64 cols (4 WMMA/kstep).
// V^T tile async-to-LDS; P tile batched f32 loads -> f16.
// ---------------------------------------------------------------------------
__global__ __launch_bounds__(128)
void ctx_gemm(const float* __restrict__ attn, const _Float16* __restrict__ Vt,
              _Float16* __restrict__ Ctx) {
  __shared__ _Float16 Ps[64 * 40];
  __shared__ _Float16 Vs[64 * 40];
  const int tid = threadIdx.x, lane = tid & 31, w = tid >> 5;
  const int bh = blockIdx.y, b = bh >> 3, h = bh & 7;
  const int qbase = blockIdx.x * 64;
  const float*    P  = attn + (size_t)bh * S_ * (size_t)S_;
  const _Float16* Vb = Vt + (size_t)bh * DK_ * S_;
  const int m = lane & 15, hl = lane >> 4;
  const int kbA = hl * 8, kbB = hl * 8;
  v8f acc[4] = {};

  for (int kt = 0; kt < S_; kt += 32) {
    __syncthreads();
    // Vt tile 64x32 f16: async memory->LDS
#pragma unroll
    for (int j = 0; j < 2; ++j) {
      const int idx = tid + j * 128;
      const int r = idx >> 2, c8 = (idx & 3) * 8;
      async_load_b128(Vs + r * 40 + c8, Vb + (size_t)r * S_ + kt + c8);
    }
    // P tile 64x32 f32 -> f16: batch loads, then convert+store
    float4 fp[4];
#pragma unroll
    for (int j = 0; j < 4; ++j) {
      const int idx = tid + j * 128;
      const int r = idx >> 3, c4 = (idx & 7) * 4;
      fp[j] = *(const float4*)(P + (size_t)(qbase + r) * S_ + kt + c4);
    }
#pragma unroll
    for (int j = 0; j < 4; ++j) {
      const int idx = tid + j * 128;
      const int r = idx >> 3, c4 = (idx & 7) * 4;
      _Float16* d = Ps + r * 40 + c4;
      d[0] = (_Float16)fp[j].x; d[1] = (_Float16)fp[j].y;
      d[2] = (_Float16)fp[j].z; d[3] = (_Float16)fp[j].w;
    }
    wait_async0();
    __syncthreads();
    const v16h a = ldsfragA(Ps + (w * 16 + m) * 40 + kbA);
#pragma unroll
    for (int t = 0; t < 4; ++t) {
      const v16h bb = ldsfragB(Vs + (t * 16 + m) * 40 + kbB);
      acc[t] = wmma_f16(a, bb, acc[t]);
    }
  }

  const int n = m;
#pragma unroll
  for (int t = 0; t < 4; ++t)
#pragma unroll
    for (int r = 0; r < 8; ++r)
      Ctx[(size_t)(b * S_ + qbase + w * 16 + r + 8 * hl) * DM_ +
          h * DK_ + t * 16 + n] = (_Float16)acc[t][r];
}

// ---------------------------------------------------------------------------
// Kernel 5: out = LN( ctx @ W_fc + input_Q ).  16 rows x full 512 cols per WG.
// Each wave: 8 col-tiles (8 WMMA / k-step). Both f16 tiles via async-to-LDS.
// ---------------------------------------------------------------------------
#define FC_SMEM (1280 + 40960 + 32768 + 512 + 512 + 64 + 64)

__global__ __launch_bounds__(128)
void fc_ln(const _Float16* __restrict__ Ctx, const _Float16* __restrict__ WfcT,
           const float* __restrict__ resid, float* __restrict__ out) {
  extern __shared__ char smem[];
  _Float16* As   = (_Float16*)smem;                            // [16][40]
  _Float16* Bs   = As + 16 * 40;                               // [512][40]
  float*    xbuf = (float*)(smem + 1280 + 40960);              // [16][512]
  float*    red  = (float*)(smem + 1280 + 40960 + 32768);      // [16][8]
  float*    red2 = red + 128;                                  // [16][8]
  float*    rowv = red2 + 128;                                 // [16] mean
  float*    rowv2= rowv + 16;                                  // [16] var

  const int tid = threadIdx.x, lane = tid & 31, w = tid >> 5;
  const int rowbase = blockIdx.x * 16;
  const int m = lane & 15, hl = lane >> 4;
  const int kbA = hl * 8, kbB = hl * 8;
  v8f acc[8] = {};

  for (int kt = 0; kt < DM_; kt += 32) {
    __syncthreads();
    if (tid < 64) {  // ctx tile 16x32 f16, async
      const int r = tid >> 2, c8 = (tid & 3) * 8;
      async_load_b128(As + r * 40 + c8,
                      Ctx + (size_t)(rowbase + r) * DM_ + kt + c8);
    }
#pragma unroll
    for (int j = 0; j < 16; ++j) {  // W_fc^T tile 512x32 f16, async
      const int idx = tid + j * 128;
      const int r = idx >> 2, c8 = (idx & 3) * 8;
      async_load_b128(Bs + r * 40 + c8,
                      WfcT + (size_t)r * DM_ + kt + c8);
    }
    wait_async0();
    __syncthreads();
    const v16h a = ldsfragA(As + m * 40 + kbA);
#pragma unroll
    for (int t = 0; t < 8; ++t) {
      const v16h bb = ldsfragB(Bs + (w * 128 + t * 16 + m) * 40 + kbB);
      acc[t] = wmma_f16(a, bb, acc[t]);
    }
  }

  // GEMM result + residual -> LDS
  const int n = m;
#pragma unroll
  for (int t = 0; t < 8; ++t)
#pragma unroll
    for (int r = 0; r < 8; ++r) {
      const int lr = r + 8 * hl;
      const int c  = w * 128 + t * 16 + n;
      xbuf[lr * DM_ + c] =
          acc[t][r] + resid[(size_t)(rowbase + lr) * DM_ + c];
    }
  __syncthreads();

  // LayerNorm: 8 threads per row
  const int row = tid >> 3, sub = tid & 7;
  float s1 = 0.f, s2 = 0.f;
  for (int c = sub; c < DM_; c += 8) {
    const float x = xbuf[row * DM_ + c];
    s1 += x; s2 += x * x;
  }
  red[row * 8 + sub]  = s1;
  red2[row * 8 + sub] = s2;
  __syncthreads();
  if (sub == 0) {
    float a1 = 0.f, a2 = 0.f;
#pragma unroll
    for (int j = 0; j < 8; ++j) { a1 += red[row * 8 + j]; a2 += red2[row * 8 + j]; }
    const float mu = a1 / (float)DM_;
    rowv[row]  = mu;
    rowv2[row] = a2 / (float)DM_ - mu * mu;
  }
  __syncthreads();
  const float mu   = rowv[row];
  const float rstd = rsqrtf(rowv2[row] + 1e-5f);
  float* drow = out + (size_t)(rowbase + row) * DM_;
  for (int c = sub; c < DM_; c += 8)
    drow[c] = (xbuf[row * DM_ + c] - mu) * rstd;
}

// ---------------------------------------------------------------------------
// Host launcher. Workspace (f16, byte offsets):
//   WqT 0 | WkT 512K | WvT 1M | WfcT 1.5M | Qf 2M | Kf 10M | Vt 18M | Ctx 26M
// Total ~34 MB.
// ---------------------------------------------------------------------------
extern "C" void kernel_launch(void* const* d_in, const int* in_sizes, int n_in,
                              void* d_out, int out_size, void* d_ws, size_t ws_size,
                              hipStream_t stream) {
  (void)in_sizes; (void)n_in; (void)out_size; (void)ws_size;
  const float* inQ = (const float*)d_in[0];
  const float* inK = (const float*)d_in[1];
  const float* inV = (const float*)d_in[2];
  const unsigned char* mask = (const unsigned char*)d_in[3];  // bool mask
  const float* Wq  = (const float*)d_in[4];
  const float* Wk  = (const float*)d_in[5];
  const float* Wv  = (const float*)d_in[6];
  const float* Wfc = (const float*)d_in[7];

  char* ws = (char*)d_ws;
  _Float16* WqT  = (_Float16*)(ws + 0);
  _Float16* WkT  = (_Float16*)(ws + 524288);
  _Float16* WvT  = (_Float16*)(ws + 1048576);
  _Float16* WfcT = (_Float16*)(ws + 1572864);
  _Float16* Qf   = (_Float16*)(ws + 2097152);
  _Float16* Kf   = (_Float16*)(ws + 10485760);
  _Float16* Vt   = (_Float16*)(ws + 18874368);
  _Float16* Ctx  = (_Float16*)(ws + 27262976);

  float* out  = (float*)d_out;
  float* attn = out + (size_t)M_ * DM_;   // attn region follows out region

  cvt_weights<<<4096, 256, 0, stream>>>(Wq, Wk, Wv, Wfc, WqT);

  gemm_xw<<<dim3(8, 128), 128, 0, stream>>>(inQ, WqT, Qf, 0);
  gemm_xw<<<dim3(8, 128), 128, 0, stream>>>(inK, WkT, Kf, 0);
  gemm_xw<<<dim3(8, 128), 128, 0, stream>>>(inV, WvT, Vt, 1);

  attn_softmax<<<dim3(64, 64), 128, ATTN_SMEM, stream>>>(Qf, Kf, mask, attn);

  ctx_gemm<<<dim3(16, 64), 128, 0, stream>>>(attn, Vt, Ctx);

  fc_ln<<<512, 128, FC_SMEM, stream>>>(Ctx, WfcT, inQ, out);
}
